// MultiHeadAttention_34007551049915
// MI455X (gfx1250) — matrix operand order repaired and checked by reading, not compiled, not run on hardware
//
#include <hip/hip_runtime.h>
#include <hip/hip_bf16.h>

typedef __bf16 bf16_t;
typedef __attribute__((ext_vector_type(16))) __bf16 v16bf;
typedef __attribute__((ext_vector_type(8)))  __bf16 v8bf;
typedef __attribute__((ext_vector_type(8)))  float  v8f;

#define BATCH   2
#define S_LEN   2048
#define DMODEL  1024
#define NHEAD   16
#define DKH     64

// workspace layout, offsets in bf16 elements
#define OFF_XBF  ((size_t)0)          // 4096x1024          (4Mi)
#define OFF_WQ   ((size_t)4194304)    // 1024x1024          (1Mi)
#define OFF_WK   ((size_t)5242880)
#define OFF_WV   ((size_t)6291456)
#define OFF_WO   ((size_t)7340032)
#define OFF_Q    ((size_t)8388608)    // [B,H,S,64]         (4Mi)
#define OFF_K    ((size_t)12582912)
#define OFF_V    ((size_t)16777216)   // [B,H,64,S] (transposed)
#define OFF_O    ((size_t)20971520)   // [B*S, D] bf16      (4Mi)
#define QKV_SZ   ((size_t)4194304)

#define XS_LD 40                    // 80B row stride: 16B-aligned, 20-dword shift -> conflict-free b128
#define TILE_E (128 * XS_LD)        // one 128x32 staged tile (padded), 10240 B

static __device__ inline v8f wmma_bf16(v16bf a, v16bf b, v8f c) {
  return __builtin_amdgcn_wmma_f32_16x16x32_bf16(
      false, a, false, b, (short)0, c, false, false);
}

// ---- async global -> LDS copy of one 16B chunk per lane (ASYNCcnt-tracked) ----
static __device__ inline void async_copy16(const bf16_t* g, unsigned lds_off) {
  asm volatile("global_load_async_to_lds_b128 %0, %1, off"
               :: "v"(lds_off), "v"((unsigned long long)(uintptr_t)g)
               : "memory");
}

// stage a 128-row x 32-col bf16 tile (global stride DMODEL) into LDS (stride XS_LD)
// 256 threads: thread t copies row t>>1, 16-elem half (t&1). 2 async ops/thread.
static __device__ inline void stage_tile(const bf16_t* src, bf16_t* dst, int tid) {
  int row  = tid >> 1;
  int colh = (tid & 1) << 4;
  const bf16_t* g = src + (size_t)row * DMODEL + colh;
  unsigned l = (unsigned)(uintptr_t)(dst + row * XS_LD + colh);
  async_copy16(g,     l);
  async_copy16(g + 8, l + 16u);
}

// A fragment from LDS tile: a[i]=T[m+lane%16][i+(i>=8?8:0)+8*(lane/16)]
static __device__ inline v16bf lds_frag_a(const bf16_t* tile, int mbase, int lane) {
  int row = mbase + (lane & 15);
  int kh  = (lane >> 4) * 8;
  const v8bf* p0 = (const v8bf*)(tile + row * XS_LD + kh);
  const v8bf* p1 = (const v8bf*)(tile + row * XS_LD + 16 + kh);
  v8bf lo = *p0, hv = *p1;
  v16bf a;
#pragma unroll
  for (int i = 0; i < 8; ++i) { a[i] = lo[i]; a[i + 8] = hv[i]; }
  return a;
}

// B fragment from LDS tile of W rows: b[i]=W[nrel+lane%16][i+(i>=8?8:0)+8*(lane/16)]
static __device__ inline v16bf lds_frag_b(const bf16_t* tile, int nrel, int lane) {
  int row = nrel + (lane & 15);
  int kh  = (lane >> 4) * 8;
  const v8bf* p0 = (const v8bf*)(tile + row * XS_LD + kh);
  v8bf lo = p0[0], hv = p0[2];
  v16bf b;
#pragma unroll
  for (int i = 0; i < 8; ++i) { b[i] = lo[i]; b[i + 8] = hv[i]; }
  return b;
}

// B fragment straight from global (column-contiguous): b[i]=B[k0+i+(i>=8?8:0)+8*(lane/16)][nbase+lane%16]
static __device__ inline v16bf load_frag_b(const bf16_t* base, int ld, int nbase, int k0, int lane) {
  int col = nbase + (lane & 15);
  int kh  = (lane >> 4) * 8;
  const v8bf* p0 = (const v8bf*)(base + (size_t)col * ld + k0 + kh);
  v8bf lo = p0[0], hv = p0[2];
  v16bf b;
#pragma unroll
  for (int i = 0; i < 8; ++i) { b[i] = lo[i]; b[i + 8] = hv[i]; }
  return b;
}

// A fragment straight from global row-major
static __device__ inline v16bf load_frag_a(const bf16_t* base, int ld, int mbase, int k0, int lane) {
  int row = mbase + (lane & 15);
  int kh  = (lane >> 4) * 8;
  const v8bf* p0 = (const v8bf*)(base + (size_t)row * ld + k0 + kh);
  const v8bf* p1 = (const v8bf*)(base + (size_t)row * ld + k0 + 16 + kh);
  v8bf lo = *p0, hv = *p1;
  v16bf a;
#pragma unroll
  for (int i = 0; i < 8; ++i) { a[i] = lo[i]; a[i + 8] = hv[i]; }
  return a;
}

// triple-buffered async-LDS GEMM core: C(128x128 block) = X(128xK) * W(128xK)^T, K=1024
// 8 waves: wm = wid&3 (32-row strip), wn = wid>>2 (64-col strip). One barrier / k-step,
// async prefetch distance 2.
static __device__ inline void gemm_core(const bf16_t* Xblk, const bf16_t* Wblk,
                                        bf16_t* xs, bf16_t* wsh,  // each 3*TILE_E
                                        int tid, int lane, int wm, int wn,
                                        v8f out[8]) {
  v8f c00 = {}, c01 = {}, c02 = {}, c03 = {};
  v8f c10 = {}, c11 = {}, c12 = {}, c13 = {};

  stage_tile(Xblk,      xs,           tid);   // stage 0
  stage_tile(Wblk,      wsh,          tid);
  stage_tile(Xblk + 32, xs + TILE_E,  tid);   // stage 1
  stage_tile(Wblk + 32, wsh + TILE_E, tid);

  for (int s = 0; s < 32; ++s) {
    // stage s complete (in-order ASYNCcnt): only stage s+1's 4 ops may remain
    if (s < 31) asm volatile("s_wait_asynccnt 0x4" ::: "memory");
    else        asm volatile("s_wait_asynccnt 0x0" ::: "memory");
    __syncthreads();  // all waves have stage-s data AND are done reading buf[(s+2)%3]
    if (s + 2 < 32) {
      int nxt = ((s + 2) % 3) * TILE_E;
      int k2  = (s + 2) * 32;
      stage_tile(Xblk + k2, xs + nxt, tid);
      stage_tile(Wblk + k2, wsh + nxt, tid);
    }
    const bf16_t* xt = xs  + (s % 3) * TILE_E;
    const bf16_t* wt = wsh + (s % 3) * TILE_E;

    v16bf a0 = lds_frag_a(xt, wm * 32,      lane);
    v16bf a1 = lds_frag_a(xt, wm * 32 + 16, lane);
    v16bf b0 = lds_frag_b(wt, wn * 64,      lane);
    v16bf b1 = lds_frag_b(wt, wn * 64 + 16, lane);
    v16bf b2 = lds_frag_b(wt, wn * 64 + 32, lane);
    v16bf b3 = lds_frag_b(wt, wn * 64 + 48, lane);

    c00 = wmma_bf16(a0, b0, c00);
    c01 = wmma_bf16(a0, b1, c01);
    c02 = wmma_bf16(a0, b2, c02);
    c03 = wmma_bf16(a0, b3, c03);
    c10 = wmma_bf16(a1, b0, c10);
    c11 = wmma_bf16(a1, b1, c11);
    c12 = wmma_bf16(a1, b2, c12);
    c13 = wmma_bf16(a1, b3, c13);
  }
  out[0] = c00; out[1] = c01; out[2] = c02; out[3] = c03;
  out[4] = c10; out[5] = c11; out[6] = c12; out[7] = c13;
}

// ---------------- kernel 0: f32 -> bf16 conversion ----------------
__global__ __launch_bounds__(256) void cvt_kernel(const float* __restrict__ x,
                                                  const float* __restrict__ wq,
                                                  const float* __restrict__ wk,
                                                  const float* __restrict__ wv,
                                                  const float* __restrict__ wo,
                                                  bf16_t* __restrict__ ws) {
  size_t i = (size_t)blockIdx.x * 256 + threadIdx.x;
  const size_t NX = (size_t)4194304, NW = (size_t)1048576;
  if (i < NX) {
    ws[OFF_XBF + i] = (bf16_t)x[i];
  } else {
    size_t j = i - NX;
    size_t w = j >> 20;
    size_t r = j & (NW - 1);
    const float* src = (w == 0) ? wq : (w == 1) ? wk : (w == 2) ? wv : wo;
    ws[OFF_WQ + j] = (bf16_t)src[r];
  }
}

// ---------------- kernel 1: fused QKV projection (grid.z selects Q/K/V) ----------------
__global__ __launch_bounds__(256) void qkv_gemm(const float* __restrict__ bq,
                                                const float* __restrict__ bk,
                                                const float* __restrict__ bv,
                                                bf16_t* __restrict__ ws) {
  __shared__ __align__(16) bf16_t xs[3 * TILE_E];
  __shared__ __align__(16) bf16_t wsh[3 * TILE_E];

  int which = blockIdx.z;
  const bf16_t* X = ws + OFF_XBF;
  const bf16_t* W = ws + OFF_WQ + (size_t)which * 1048576;
  const float* bias = (which == 0) ? bq : (which == 1) ? bk : bv;
  bf16_t* dst = ws + OFF_Q + (size_t)which * QKV_SZ;

  int tid = threadIdx.x, lane = tid & 31, wid = tid >> 5;
  int wm = wid & 3, wn = wid >> 2;
  int mblk = blockIdx.x * 128;
  int nblk = blockIdx.y * 128;

  v8f acc[8];
  gemm_core(X + (size_t)mblk * DMODEL, W + (size_t)nblk * DMODEL, xs, wsh,
            tid, lane, wm, wn, acc);

  int nn = lane & 15, half = lane >> 4;
#pragma unroll
  for (int mi = 0; mi < 2; ++mi) {
#pragma unroll
    for (int t = 0; t < 4; ++t) {
      int j = nblk + wn * 64 + t * 16 + nn;
      float bval = bias[j];
      int h = j >> 6, dd = j & 63;
#pragma unroll
      for (int r = 0; r < 8; ++r) {
        int i = mblk + wm * 32 + mi * 16 + r + 8 * half;
        int bb = i >> 11, s = i & 2047;
        float v = acc[mi * 4 + t][r] + bval;
        if (which == 2)  // V transposed: [B,H,64,S]
          dst[(((size_t)bb * NHEAD + h) * DKH + dd) * S_LEN + s] = (bf16_t)v;
        else             // Q/K: [B,H,S,64]
          dst[(((size_t)bb * NHEAD + h) * S_LEN + s) * DKH + dd] = (bf16_t)v;
      }
    }
  }
}

// ---------------- kernel 2: causal flash attention ----------------
__global__ __launch_bounds__(256) void flash_attn(bf16_t* __restrict__ ws) {
  const bf16_t* Q = ws + OFF_Q;
  const bf16_t* K = ws + OFF_K;
  const bf16_t* V = ws + OFF_V;  // [B,H,64,S]
  bf16_t* O = ws + OFF_O;        // [B*S, D]

  __shared__ __align__(16) bf16_t pbuf[8 * 16 * 32];  // per-wave 16x32 P tile

  int lane = threadIdx.x & 31, wid = threadIdx.x >> 5;
  int h = blockIdx.y, b = blockIdx.z;
  int qs = blockIdx.x * 128 + wid * 16;

  const bf16_t* Qh = Q + ((size_t)b * NHEAD + h) * S_LEN * DKH;
  const bf16_t* Kh = K + ((size_t)b * NHEAD + h) * S_LEN * DKH;
  const bf16_t* Vh = V + ((size_t)b * NHEAD + h) * DKH * S_LEN;

  int nn = lane & 15, half = lane >> 4;

  v16bf aq0 = load_frag_a(Qh, DKH, qs, 0, lane);
  v16bf aq1 = load_frag_a(Qh, DKH, qs, 32, lane);

  v8f o0 = {}, o1 = {}, o2 = {}, o3 = {};
  float run_max[8], run_sum[8];
#pragma unroll
  for (int r = 0; r < 8; ++r) { run_max[r] = -1e30f; run_sum[r] = 0.0f; }

  bf16_t* pb = pbuf + (size_t)wid * (16 * 32);
  const float scl = 0.125f;  // 1/sqrt(64)

  for (int jo = 0; jo < qs + 16; jo += 32) {
    v8f s0 = {}, s1 = {};
    {
      v16bf bk = load_frag_b(Kh, DKH, jo, 0, lane);
      s0 = wmma_bf16(aq0, bk, s0);
      bk = load_frag_b(Kh, DKH, jo, 32, lane);
      s0 = wmma_bf16(aq1, bk, s0);
      bk = load_frag_b(Kh, DKH, jo + 16, 0, lane);
      s1 = wmma_bf16(aq0, bk, s1);
      bk = load_frag_b(Kh, DKH, jo + 16, 32, lane);
      s1 = wmma_bf16(aq1, bk, s1);
    }
#pragma unroll
    for (int r = 0; r < 8; ++r) {
      int qrow = qs + r + 8 * half;
      int kv0 = jo + nn, kv1 = jo + 16 + nn;
      float v0 = s0[r] * scl; if (kv0 > qrow) v0 = -1e30f;
      float v1 = s1[r] * scl; if (kv1 > qrow) v1 = -1e30f;
      float mx = fmaxf(v0, v1);
#pragma unroll
      for (int msk = 1; msk < 16; msk <<= 1) mx = fmaxf(mx, __shfl_xor(mx, msk, 16));
      float newm = fmaxf(run_max[r], mx);
      float corr = __expf(run_max[r] - newm);
      run_max[r] = newm;
      float p0 = __expf(v0 - newm);
      float p1 = __expf(v1 - newm);
      float ps = p0 + p1;
#pragma unroll
      for (int msk = 1; msk < 16; msk <<= 1) ps += __shfl_xor(ps, msk, 16);
      run_sum[r] = run_sum[r] * corr + ps;
      o0[r] *= corr; o1[r] *= corr; o2[r] *= corr; o3[r] *= corr;
      int m = r + 8 * half;
      pb[m * 32 + nn]      = (bf16_t)p0;
      pb[m * 32 + 16 + nn] = (bf16_t)p1;
    }
    asm volatile("s_wait_dscnt 0x0" ::: "memory");  // same-wave LDS RAW

    v16bf ap;
    {
      int m = nn, kh = half * 8;
      const v8bf* p0 = (const v8bf*)(pb + m * 32 + kh);
      const v8bf* p1 = (const v8bf*)(pb + m * 32 + 16 + kh);
      v8bf lo = *p0, hv = *p1;
#pragma unroll
      for (int i = 0; i < 8; ++i) { ap[i] = lo[i]; ap[i + 8] = hv[i]; }
    }
    v16bf bv;
    bv = load_frag_b(Vh, S_LEN, 0,  jo, lane); o0 = wmma_bf16(ap, bv, o0);
    bv = load_frag_b(Vh, S_LEN, 16, jo, lane); o1 = wmma_bf16(ap, bv, o1);
    bv = load_frag_b(Vh, S_LEN, 32, jo, lane); o2 = wmma_bf16(ap, bv, o2);
    bv = load_frag_b(Vh, S_LEN, 48, jo, lane); o3 = wmma_bf16(ap, bv, o3);
  }

#pragma unroll
  for (int r = 0; r < 8; ++r) {
    float inv = 1.0f / run_sum[r];
    int srow = qs + r + 8 * half;
    size_t row = ((size_t)b * S_LEN + srow) * DMODEL + (size_t)h * DKH;
    O[row + 0  + nn] = (bf16_t)(o0[r] * inv);
    O[row + 16 + nn] = (bf16_t)(o1[r] * inv);
    O[row + 32 + nn] = (bf16_t)(o2[r] * inv);
    O[row + 48 + nn] = (bf16_t)(o3[r] * inv);
  }
}

// ---------------- kernel 3: output projection -> f32 ----------------
__global__ __launch_bounds__(256) void out_gemm(const float* __restrict__ bo,
                                                const bf16_t* __restrict__ wsc,
                                                float* __restrict__ out) {
  __shared__ __align__(16) bf16_t xs[3 * TILE_E];
  __shared__ __align__(16) bf16_t wsh[3 * TILE_E];

  const bf16_t* A = wsc + OFF_O;
  const bf16_t* W = wsc + OFF_WO;

  int tid = threadIdx.x, lane = tid & 31, wid = tid >> 5;
  int wm = wid & 3, wn = wid >> 2;
  int mblk = blockIdx.x * 128;
  int nblk = blockIdx.y * 128;

  v8f acc[8];
  gemm_core(A + (size_t)mblk * DMODEL, W + (size_t)nblk * DMODEL, xs, wsh,
            tid, lane, wm, wn, acc);

  int nn = lane & 15, half = lane >> 4;
#pragma unroll
  for (int mi = 0; mi < 2; ++mi) {
#pragma unroll
    for (int t = 0; t < 4; ++t) {
      int j = nblk + wn * 64 + t * 16 + nn;
      float bval = bo[j];
#pragma unroll
      for (int r = 0; r < 8; ++r) {
        int i = mblk + wm * 32 + mi * 16 + r + 8 * half;
        out[(size_t)i * DMODEL + j] = acc[mi * 4 + t][r] + bval;
      }
    }
  }
}

extern "C" void kernel_launch(void* const* d_in, const int* in_sizes, int n_in,
                              void* d_out, int out_size, void* d_ws, size_t ws_size,
                              hipStream_t stream) {
  const float* x  = (const float*)d_in[0];
  const float* Wq = (const float*)d_in[1];
  const float* bq = (const float*)d_in[2];
  const float* Wk = (const float*)d_in[3];
  const float* bk = (const float*)d_in[4];
  const float* Wv = (const float*)d_in[5];
  const float* bv = (const float*)d_in[6];
  const float* Wo = (const float*)d_in[7];
  const float* bo = (const float*)d_in[8];
  // d_in[9] = causal mask, handled analytically in flash_attn

  bf16_t* ws = (bf16_t*)d_ws;
  float* out = (float*)d_out;

  cvt_kernel<<<32768, 256, 0, stream>>>(x, Wq, Wk, Wv, Wo, ws);

  dim3 g1(32, 8, 3);   // M(4096/128) x N(1024/128) x {Q,K,V}
  qkv_gemm<<<g1, 256, 0, stream>>>(bq, bk, bv, ws);

  dim3 g2(16, 16, 2);  // q-tiles(2048/128) x heads x batch
  flash_attn<<<g2, 256, 0, stream>>>(ws);

  dim3 g3(32, 8);      // M x N
  out_gemm<<<g3, 256, 0, stream>>>(bo, ws, out);
}
